// SimpleRecurrentSequentialModel_12799002542578
// MI455X (gfx1250) — compile-verified, matrix operand order checked by reading, not run on hardware
//
#include <hip/hip_runtime.h>

// SimpleRecurrentSequentialModel for MI455X (gfx1250, wave32, WMMA bf16 + TDM)
//
// Pipeline (all on `stream`):
//  1) cvt_bf16       : Wh, Wx, Wo(fp32) -> bf16 (Wo zero-padded to 50304 rows)
//  2) embed_gather   : x[rm=l*4+b][d] = bf16(emb[seq[b][l]][d])
//  3) xproj_gemm     : xprojF[rm][e] = x @ Wx^T + bx   (WMMA bf16, f32 out)
//  4) scan_kernel    : 1 workgroup, 16 waves; Wh resident in VGPRs as WMMA
//                      B-fragments (256 VGPRs/lane); h in LDS; 1024 steps of
//                      relu(h@Wh^T + bh + xt); emits hs as bf16
//  5) out_gemm       : out[b][l][v] = hs @ Wo^T + bo   (WMMA bf16, f32 out)
//
// B-tile staging into LDS uses the Tensor Data Mover (tensor_load_to_lds)
// with hardware row padding (4 dwords after every 256 dwords -> 1040B rows,
// bank-conflict free), falling back to load_b128+ds_store_b128 staging.

#define B_    4
#define L_    1024
#define V_    50257
#define D_    512
#define VPAD  50304            // 393 * 128
#define M_    4096             // B_ * L_
#define LDSB_STRIDE 520        // bf16 elems per staged row (bank-conflict pad)
#define HSTRIDE     520

typedef __attribute__((ext_vector_type(16))) __bf16 v16bf;
typedef __attribute__((ext_vector_type(8)))  float  v8f;

union BFrag { v16bf v; uint4 q[2]; };
union FAcc  { v8f  v; float f[8]; };

static __device__ __forceinline__ __bf16 to_bf16(float x) { return (__bf16)x; }

#define WMMA_BF16(a, b, c) \
  __builtin_amdgcn_wmma_f32_16x16x32_bf16(false, (a), false, (b), (short)0, (c), false, false)

// ---------------------------------------------------------------- TDM staging
#if defined(__gfx1250__) && __has_builtin(__builtin_amdgcn_tensor_load_to_lds) && \
    __has_builtin(__builtin_amdgcn_s_wait_tensorcnt)
#define USE_TDM 1
#else
#define USE_TDM 0
#endif

#if USE_TDM
typedef unsigned int v4u_t __attribute__((ext_vector_type(4)));
typedef int          v8i_t __attribute__((ext_vector_type(8)));
typedef int          v4i_t __attribute__((ext_vector_type(4)));
#define AS3_ __attribute__((address_space(3)))

// DMA a 128-row x 512-col bf16 tile (row stride 512 elems in memory) into LDS
// at lds_byte_off, inserting 4 dwords of padding after every 256 dwords so the
// LDS row stride becomes 1040 bytes (= LDSB_STRIDE bf16).  D# layout per
// CDNA5 ISA.
static __device__ __forceinline__ void tdm_load_rows128(const __bf16* gsrc,
                                                        unsigned lds_byte_off,
                                                        unsigned tensor_rows) {
  union { v4u_t v; unsigned w[4]; } g0;
  union { v8i_t v; unsigned long long q[4]; } g1;
  union { v4i_t v; int w[4]; } gz;
  unsigned long long ga = (unsigned long long)(size_t)gsrc;
  g0.w[0] = 1u;                                  // count=1 (valid descriptor)
  g0.w[1] = lds_byte_off;                        // lds_addr
  g0.w[2] = (unsigned)ga;                        // global_addr[31:0]
  g0.w[3] = (unsigned)(ga >> 32) | (2u << 30);   // global_addr[56:32] | type=2
  g1.q[0] = (unsigned long long)((1u << 16)      // data_size = 2 bytes
                               | (1u << 20)      // pad_enable
                               | (7u << 22)      // pad_interval = 256 dwords
                               | (3u << 25))     // pad_amount   = 4 dwords
          | (512ull << 48);                      // tensor_dim0 = 512
  g1.q[1] = ((unsigned long long)tensor_rows << 16)  // tensor_dim1
          | (512ull << 48);                      // tile_dim0 = 512
  g1.q[2] = 128ull                               // tile_dim1 = 128 rows
          | (512ull << 32);                      // tensor_dim0_stride = 512
  g1.q[3] = (512ull << 16);                      // tensor_dim1_stride = 512
  gz.w[0] = gz.w[1] = gz.w[2] = gz.w[3] = 0;
#if __has_include(<hip/amd_detail/amd_gfx1250_TDM.h>)
  union { v8i_t v; int w[8]; } gz8;
#pragma unroll
  for (int i = 0; i < 8; i++) gz8.w[i] = 0;
  __builtin_amdgcn_tensor_load_to_lds(g0.v, g1.v, gz.v, gz.v, gz8.v, 0);
#else
  __builtin_amdgcn_tensor_load_to_lds(g0.v, g1.v, gz.v, gz.v, 0);
#endif
}
#endif  // USE_TDM

// ---------------------------------------------------------------- converts
__global__ void cvt_bf16(const float* __restrict__ src, __bf16* __restrict__ dst,
                         int rows_valid, int total) {
  int i = blockIdx.x * blockDim.x + threadIdx.x;
  if (i >= total) return;
  int r = i >> 9;                       // 512 cols per row
  float v = 0.0f;
  if (r < rows_valid) v = src[i];
  dst[i] = to_bf16(v);
}

__global__ void embed_gather(const int* __restrict__ seq, const float* __restrict__ emb,
                             __bf16* __restrict__ xBf) {
  int i = blockIdx.x * blockDim.x + threadIdx.x;   // < 4096*512
  int d  = i & 511;
  int rm = i >> 9;                                  // rm = l*4 + b
  int b = rm & 3, l = rm >> 2;
  int tok = seq[b * L_ + l];
  xBf[i] = to_bf16(emb[(size_t)tok * D_ + d]);
}

// ---------------------------------------------------------------- xproj GEMM
// M=4096, N=512, K=512.  Block: 128M x 128N, 8 waves, wave w -> m-tile w,
// 8 n-tiles per wave.  B (Wx rows) staged in LDS via TDM.
__global__ __launch_bounds__(256) void xproj_gemm(const __bf16* __restrict__ xBf,
                                                  const __bf16* __restrict__ WxBf,
                                                  const float* __restrict__ bx,
                                                  float* __restrict__ xprojF) {
  __shared__ __bf16 Bs[128 * LDSB_STRIDE];
  const int tid  = threadIdx.x;
  const int lane = tid & 31, wave = tid >> 5;
  const int half = lane >> 4, lm = lane & 15;
  const int nb = blockIdx.x;           // 128-wide N block
  const int mb = blockIdx.y;           // 128-tall M block

#if USE_TDM
  if (wave == 0) {
    tdm_load_rows128(WxBf + (size_t)nb * 128 * D_,
                     (unsigned)(size_t)(AS3_ char*)Bs, 512u);
    __builtin_amdgcn_s_wait_tensorcnt((short)0);
  }
#else
  for (int i = tid; i < 128 * 32; i += 256) {       // 128 rows x 32 16B-chunks
    int row = i >> 5, c16 = i & 31;
    uint4 dql = *(const uint4*)(WxBf + (size_t)(nb * 128 + row) * D_ + c16 * 8);
    *(uint4*)(Bs + row * LDSB_STRIDE + c16 * 8) = dql;
  }
#endif
  __syncthreads();

  FAcc c[8];
#pragma unroll
  for (int nt = 0; nt < 8; nt++)
#pragma unroll
    for (int r = 0; r < 8; r++) c[nt].f[r] = 0.0f;

  const int mrow = mb * 128 + wave * 16 + lm;       // A row for this lane
  const __bf16* aBase = xBf + (size_t)mrow * D_;
#pragma unroll
  for (int kf = 0; kf < 16; kf++) {
    BFrag a;
    a.q[0] = *(const uint4*)(aBase + kf * 32 + half * 8);
    a.q[1] = *(const uint4*)(aBase + kf * 32 + 16 + half * 8);
#pragma unroll
    for (int nt = 0; nt < 8; nt++) {
      BFrag bfr;
      const __bf16* bp = Bs + (nt * 16 + lm) * LDSB_STRIDE + kf * 32 + half * 8;
      bfr.q[0] = *(const uint4*)bp;
      bfr.q[1] = *(const uint4*)(bp + 16);
      c[nt].v = WMMA_BF16(a.v, bfr.v, c[nt].v);
    }
  }

#pragma unroll
  for (int nt = 0; nt < 8; nt++) {
    int e = nb * 128 + nt * 16 + lm;
    float bxv = bx[e];
#pragma unroll
    for (int r = 0; r < 8; r++) {
      int row = mb * 128 + wave * 16 + r + half * 8;   // D row = r + 8*half
      xprojF[(size_t)row * D_ + e] = c[nt].f[r] + bxv;
    }
  }
}

// ---------------------------------------------------------------- scan
// One workgroup, 512 threads = 16 waves.  Wave w owns h_new columns
// [32w, 32w+32) as two 16x16 tiles.  Wh B-fragments (2 tiles x 16 kfrag x
// 8 VGPR = 256 VGPRs/lane) live in registers for all 1024 steps.
// h (16 x 512 bf16, rows 4..15 always zero) lives in LDS with padded stride.
__global__ __launch_bounds__(512, 1) void scan_kernel(const __bf16* __restrict__ WhBf,
                                                      const float* __restrict__ bh,
                                                      const float* __restrict__ xprojF,
                                                      __bf16* __restrict__ hsBf) {
  __shared__ __bf16 hL[16 * HSTRIDE];
  const int tid  = threadIdx.x;
  const int lane = tid & 31, w = tid >> 5;
  const int half = lane >> 4, lm = lane & 15;

  for (int i = tid; i < (16 * HSTRIDE) / 2; i += 512)  // zero h0 (dwords)
    ((unsigned int*)hL)[i] = 0u;

  BFrag Bf[2][16];
#pragma unroll
  for (int tt = 0; tt < 2; tt++) {
    int n = w * 32 + tt * 16 + lm;                      // Wh row e = B column
    const __bf16* bbase = WhBf + (size_t)n * D_;
#pragma unroll
    for (int kf = 0; kf < 16; kf++) {
      Bf[tt][kf].q[0] = *(const uint4*)(bbase + kf * 32 + half * 8);
      Bf[tt][kf].q[1] = *(const uint4*)(bbase + kf * 32 + 16 + half * 8);
    }
  }
  float bhv0 = bh[w * 32 + lm];
  float bhv1 = bh[w * 32 + 16 + lm];
  __syncthreads();

  for (int t = 0; t < L_; t++) {
    FAcc c0, c1;
#pragma unroll
    for (int r = 0; r < 8; r++) { c0.f[r] = 0.0f; c1.f[r] = 0.0f; }

#pragma unroll
    for (int kf = 0; kf < 16; kf++) {
      BFrag a;
      if (lm < 4) {                       // rows m>=4 of h are always zero
        const __bf16* ap = hL + lm * HSTRIDE + kf * 32 + half * 8;
        a.q[0] = *(const uint4*)ap;
        a.q[1] = *(const uint4*)(ap + 16);
      } else {
        a.q[0] = make_uint4(0, 0, 0, 0);
        a.q[1] = make_uint4(0, 0, 0, 0);
      }
      c0.v = WMMA_BF16(a.v, Bf[0][kf].v, c0.v);
      c1.v = WMMA_BF16(a.v, Bf[1][kf].v, c1.v);
    }
    __syncthreads();                      // all h reads done before overwrite

    if (half == 0) {                      // D rows r (= batch m) for r<4
#pragma unroll
      for (int r = 0; r < 4; r++) {
        int n0 = w * 32 + lm;
        int n1 = n0 + 16;
        const float* xp = xprojF + (size_t)(t * 4 + r) * D_;
        float v0 = fmaxf(c0.f[r] + bhv0 + xp[n0], 0.0f);
        float v1 = fmaxf(c1.f[r] + bhv1 + xp[n1], 0.0f);
        __bf16 b0 = to_bf16(v0), b1 = to_bf16(v1);
        hL[r * HSTRIDE + n0] = b0;
        hL[r * HSTRIDE + n1] = b1;
        hsBf[(size_t)(t * 4 + r) * D_ + n0] = b0;
        hsBf[(size_t)(t * 4 + r) * D_ + n1] = b1;
      }
    }
    __syncthreads();
  }
}

// ---------------------------------------------------------------- output GEMM
// M=4096, N=50304(pad of 50257), K=512.  Block: 128M x 128N, Wo block staged
// in LDS via TDM.  Epilogue adds bo and scatters to out[b][l][v] (fp32).
__global__ __launch_bounds__(256) void out_gemm(const __bf16* __restrict__ hsBf,
                                                const __bf16* __restrict__ WoBf,
                                                const float* __restrict__ bo,
                                                float* __restrict__ out) {
  __shared__ __bf16 Bs[128 * LDSB_STRIDE];
  const int tid  = threadIdx.x;
  const int lane = tid & 31, wave = tid >> 5;
  const int half = lane >> 4, lm = lane & 15;
  const int nb = blockIdx.x;           // 128-wide V block (393 blocks)
  const int mb = blockIdx.y;           // 128-tall M block (32 blocks)

#if USE_TDM
  if (wave == 0) {
    tdm_load_rows128(WoBf + (size_t)nb * 128 * D_,
                     (unsigned)(size_t)(AS3_ char*)Bs, (unsigned)VPAD);
    __builtin_amdgcn_s_wait_tensorcnt((short)0);
  }
#else
  for (int i = tid; i < 128 * 32; i += 256) {
    int row = i >> 5, c16 = i & 31;
    uint4 dql = *(const uint4*)(WoBf + (size_t)(nb * 128 + row) * D_ + c16 * 8);
    *(uint4*)(Bs + row * LDSB_STRIDE + c16 * 8) = dql;
  }
#endif
  __syncthreads();

  FAcc c[8];
#pragma unroll
  for (int nt = 0; nt < 8; nt++)
#pragma unroll
    for (int r = 0; r < 8; r++) c[nt].f[r] = 0.0f;

  const int mrow = mb * 128 + wave * 16 + lm;
  const __bf16* aBase = hsBf + (size_t)mrow * D_;
#pragma unroll
  for (int kf = 0; kf < 16; kf++) {
    BFrag a;
    a.q[0] = *(const uint4*)(aBase + kf * 32 + half * 8);
    a.q[1] = *(const uint4*)(aBase + kf * 32 + 16 + half * 8);
#pragma unroll
    for (int nt = 0; nt < 8; nt++) {
      BFrag bfr;
      const __bf16* bp = Bs + (nt * 16 + lm) * LDSB_STRIDE + kf * 32 + half * 8;
      bfr.q[0] = *(const uint4*)bp;
      bfr.q[1] = *(const uint4*)(bp + 16);
      c[nt].v = WMMA_BF16(a.v, bfr.v, c[nt].v);
    }
  }

#pragma unroll
  for (int nt = 0; nt < 8; nt++) {
    int v = nb * 128 + nt * 16 + lm;
    if (v < V_) {
      float bov = bo[v];
#pragma unroll
      for (int r = 0; r < 8; r++) {
        int rm = mb * 128 + wave * 16 + r + half * 8;   // rm = l*4 + b
        int b = rm & 3, l = rm >> 2;
        out[((size_t)b * L_ + l) * V_ + v] = c[nt].f[r] + bov;
      }
    }
  }
}

// ---------------------------------------------------------------- launch
extern "C" void kernel_launch(void* const* d_in, const int* in_sizes, int n_in,
                              void* d_out, int out_size, void* d_ws, size_t ws_size,
                              hipStream_t stream) {
  const int*   seq = (const int*)  d_in[0];
  const float* emb = (const float*)d_in[1];
  const float* Wh  = (const float*)d_in[2];
  const float* bh  = (const float*)d_in[3];
  const float* Wx  = (const float*)d_in[4];
  const float* bx  = (const float*)d_in[5];
  const float* Wo  = (const float*)d_in[6];
  const float* bo  = (const float*)d_in[7];
  float* out = (float*)d_out;

  char* ws = (char*)d_ws;
  size_t off = 0;
  __bf16* WhBf = (__bf16*)(ws + off); off += (size_t)512 * 512 * 2;
  __bf16* WxBf = (__bf16*)(ws + off); off += (size_t)512 * 512 * 2;
  __bf16* WoBf = (__bf16*)(ws + off); off += (size_t)VPAD * 512 * 2;
  __bf16* xBf  = (__bf16*)(ws + off); off += (size_t)M_ * 512 * 2;
  float*  xprojF = (float*)(ws + off); off += (size_t)M_ * 512 * 4;
  __bf16* hsBf = (__bf16*)(ws + off); off += (size_t)M_ * 512 * 2;
  (void)off; (void)ws_size; (void)in_sizes; (void)n_in; (void)out_size;

  cvt_bf16<<<(512 * 512 + 255) / 256, 256, 0, stream>>>(Wh, WhBf, 512, 512 * 512);
  cvt_bf16<<<(512 * 512 + 255) / 256, 256, 0, stream>>>(Wx, WxBf, 512, 512 * 512);
  cvt_bf16<<<(VPAD * 512 + 255) / 256, 256, 0, stream>>>(Wo, WoBf, V_, VPAD * 512);
  embed_gather<<<(M_ * 512) / 256, 256, 0, stream>>>(seq, emb, xBf);
  xproj_gemm<<<dim3(4, 32), 256, 0, stream>>>(xBf, WxBf, bx, xprojF);
  scan_kernel<<<1, 512, 0, stream>>>(WhBf, bh, xprojF, hsBf);
  out_gemm<<<dim3(VPAD / 128, 32), 256, 0, stream>>>(hsBf, WoBf, bo, out);
}
